// Part3Loss_37615323578676
// MI455X (gfx1250) — compile-verified, matrix operand order checked by reading, not run on hardware
//
#include <hip/hip_runtime.h>
#include <math.h>

typedef __attribute__((ext_vector_type(16))) _Float16 v16h;
typedef __attribute__((ext_vector_type(8)))  float    v8f;

#if defined(__gfx1250__) && __has_builtin(__builtin_amdgcn_tensor_load_to_lds)
#define HAVE_TDM 1
typedef __attribute__((ext_vector_type(4))) unsigned int uint32x4;
typedef __attribute__((ext_vector_type(8))) int          int32x8;
typedef __attribute__((ext_vector_type(4))) int          int32x4;
#else
#define HAVE_TDM 0
#endif

#define BB 64
#define KK 4000
#define GG 200
#define GPAD 208            // 13 groups of 16 pairs for WMMA
#define CC 91
#define EPSF 1e-9f
#define NTHREADS 256
#define NWAVES 8
#define ROWWORDS 125        // 4000 / 32

__device__ __forceinline__ float areaOf(float x0, float y0, float x1, float y1) {
    return fmaxf(x1 - x0, 0.f) * fmaxf(y1 - y0, 0.f);
}

__device__ __forceinline__ float iouPG(float4 p, float4 g, float ap, float ag) {
    float tlx = fmaxf(p.x, g.x), tly = fmaxf(p.y, g.y);
    float brx = fminf(p.z, g.z), bry = fminf(p.w, g.w);
    float w = fmaxf(brx - tlx, 0.f), h = fmaxf(bry - tly, 0.f);
    float inter = w * h;
    float uni = fmaxf(ap + ag - inter, EPSF);
    return inter / uni;
}

__global__ void __launch_bounds__(NTHREADS)
loss_init_kernel(float* ws) {
    if (threadIdx.x < 4) ws[threadIdx.x] = 0.f;
}

__global__ void __launch_bounds__(NTHREADS)
loss_batch_kernel(const float* __restrict__ pred_boxes,
                  const float* __restrict__ pred_obj,
                  const float* __restrict__ pred_cls,
                  const float* __restrict__ gt_boxes,
                  const int*   __restrict__ gt_labels,
                  const unsigned char* __restrict__ gt_mask,
                  float* __restrict__ ws) {
    __shared__ float4   sGt[GPAD];
    __shared__ float    sAreaG[GPAD];
    __shared__ float    sColMax[GPAD];
    __shared__ int      sColArg[GPAD];
    __shared__ unsigned sRowAlive[ROWWORDS];
    __shared__ int      sPs[GPAD];
    __shared__ int      sGs[GPAD];
    __shared__ int      sVs[GPAD];
    __shared__ float    sMaxl[GPAD];
    __shared__ float    sLabLogit[GPAD];
    __shared__ float    sSumexp[GPAD];
    __shared__ float    sWBestV[NWAVES];
    __shared__ int      sWBestR[NWAVES];
    __shared__ int      sAffList[GPAD];
    __shared__ int      sAffCnt;
    __shared__ float    sSelVal;
    __shared__ int      sSelP, sSelG;
    __shared__ float    sLbox, sLcls, sObjPos, sObjNeg;
    __shared__ int      sN;

    const int b    = blockIdx.x;
    const int tid  = threadIdx.x;
    const int lane = tid & 31;
    const int wv   = tid >> 5;

    const float4* pb = reinterpret_cast<const float4*>(pred_boxes) + (size_t)b * KK;
    const float4* gb = reinterpret_cast<const float4*>(gt_boxes)   + (size_t)b * GG;

    // ---- Phase 0: init shared state; gt tile moved via Tensor Data Mover ----
#if HAVE_TDM
    if (wv == 0) {
        // D# group 0: count=1, lds_addr, 57-bit global addr, type=2 ("image")
        unsigned long long ga = (unsigned long long)(const void*)gb;
        unsigned ldsAddr = (unsigned)(size_t)(const void*)&sGt[0];  // low 32b of generic = LDS byte addr
        uint32x4 g0;
        g0[0] = 1u;                                                // count=1, user mode
        g0[1] = ldsAddr;
        g0[2] = (unsigned)(ga & 0xFFFFFFFFull);
        g0[3] = ((unsigned)(ga >> 32) & 0x01FFFFFFu) | 0x80000000u; // addr[56:32] | type=2<<30
        // D# group 1: data_size=8B, 1 row of 400 elements (3200B), stride 400
        int32x8 g1;
        g1[0] = 0x00030000;           // workgroup_mask=0, data_size=3 (8B)
        g1[1] = (int)(400u << 16);    // tensor_dim0 = 400 (lo16 at bits 63:48)
        g1[2] = (int)(1u << 16);      // tensor_dim0 hi=0 | tensor_dim1 = 1
        g1[3] = (int)(400u << 16);    // tensor_dim1 hi=0 | tile_dim0 = 400
        g1[4] = 1;                    // tile_dim1 = 1, tile_dim2 = 0
        g1[5] = 400;                  // tensor_dim0_stride lo32
        g1[6] = (int)(400u << 16);    // stride0 hi=0 | tensor_dim1_stride lo16
        g1[7] = 0;
        int32x4 gz4 = {0, 0, 0, 0};            // groups 2/3 unused (<=2D tensor)
        int32x8 gz8 = {0, 0, 0, 0, 0, 0, 0, 0};
        __builtin_amdgcn_tensor_load_to_lds(g0, g1, gz4, gz4, gz8, 0);  // 6-arg (clang-23) form
        __builtin_amdgcn_s_wait_tensorcnt(0);
    }
#endif
    for (int i = tid; i < GPAD; i += NTHREADS) {
#if HAVE_TDM
        if (i >= GG) sGt[i] = make_float4(0.f, 0.f, 0.f, 0.f);
#else
        sGt[i] = (i < GG) ? gb[i] : make_float4(0.f, 0.f, 0.f, 0.f);
#endif
        sPs[i] = 0; sGs[i] = 0; sVs[i] = 0;
        sMaxl[i] = 0.f; sLabLogit[i] = 0.f; sSumexp[i] = 1.f;
    }
    for (int i = tid; i < ROWWORDS; i += NTHREADS) sRowAlive[i] = 0xFFFFFFFFu;
    if (tid == 0) {
        sAffCnt = 0; sN = 0;
        sLbox = 0.f; sLcls = 0.f; sObjPos = 0.f; sObjNeg = 0.f;
    }
    __syncthreads();                 // TDM complete (tensorcnt waited) + LDS init visible
    for (int i = tid; i < GPAD; i += NTHREADS) {
        float4 g = sGt[i];
        sAreaG[i] = areaOf(g.x, g.y, g.z, g.w);
    }
    __syncthreads();

    // ---- Phase 1: initial per-column max IoU (thread per gt column) ----
    if (tid < GPAD) {
        bool alive = (tid < GG) && (gt_mask[(size_t)b * GG + tid] != 0);
        float best = -1.f; int bArg = 0;
        if (alive) {
            float4 g = sGt[tid]; float ag = sAreaG[tid];
            for (int r = 0; r < KK; ++r) {
                __builtin_prefetch(pb + r + 32, 0, 0);   // global_prefetch_b8, L2-resident stream
                float4 p = pb[r];
                float ap = areaOf(p.x, p.y, p.z, p.w);
                float v = iouPG(p, g, ap, ag);
                if (v > best) { best = v; bArg = r; }    // strict > : smallest row on tie
            }
        }
        sColMax[tid] = alive ? best : -1.f;
        sColArg[tid] = bArg;
    }
    __syncthreads();

    // ---- Phase 2: greedy matching (incremental column-max maintenance) ----
    for (int step = 0; step < GG; ++step) {
        if (tid < 32) {
            float bv = -2.f; unsigned bkey = 0xFFFFFFFFu;
            for (int g2 = lane; g2 < GPAD; g2 += 32) {
                float v = sColMax[g2];
                unsigned key = (unsigned)sColArg[g2] * 200u + (unsigned)g2; // flat p*G+g
                if (v > bv || (v == bv && key < bkey)) { bv = v; bkey = key; }
            }
            for (int off = 16; off; off >>= 1) {
                float ov = __shfl_xor(bv, off);
                unsigned ok = (unsigned)__shfl_xor((int)bkey, off);
                if (ov > bv || (ov == bv && ok < bkey)) { bv = ov; bkey = ok; }
            }
            if (lane == 0) { sSelVal = bv; sSelP = (int)(bkey / 200u); sSelG = (int)(bkey % 200u); }
        }
        __syncthreads();
        float mval = sSelVal;
        int p = sSelP, g = sSelG;
        if (mval < 0.f) break;  // uniform: once invalid, stays invalid (reference scan semantics)
        if (tid == 0) {
            sVs[step] = 1; sPs[step] = p; sGs[step] = g; sN++;
            sColMax[g] = -2.f;
            sRowAlive[p >> 5] &= ~(1u << (p & 31));
            sAffCnt = 0;
        }
        __syncthreads();
        if (tid < GPAD) {
            if (sColMax[tid] >= 0.f && sColArg[tid] == p) {
                int s2 = atomicAdd(&sAffCnt, 1);
                sAffList[s2] = tid;
            }
        }
        __syncthreads();
        int nAff = sAffCnt;
        for (int a = 0; a < nAff; ++a) {
            int g2 = sAffList[a];
            float4 gbx = sGt[g2]; float ag = sAreaG[g2];
            float bv = -1.f; int br = KK;
            for (int r = tid; r < KK; r += NTHREADS) {
                __builtin_prefetch(pb + r + NTHREADS, 0, 0);
                if (sRowAlive[r >> 5] & (1u << (r & 31))) {
                    float4 pbx = pb[r];
                    float ap = areaOf(pbx.x, pbx.y, pbx.z, pbx.w);
                    float v = iouPG(pbx, gbx, ap, ag);
                    if (v > bv) { bv = v; br = r; }
                }
            }
            for (int off = 16; off; off >>= 1) {
                float ov = __shfl_xor(bv, off);
                int orr = __shfl_xor(br, off);
                if (ov > bv || (ov == bv && orr < br)) { bv = ov; br = orr; }
            }
            if (lane == 0) { sWBestV[wv] = bv; sWBestR[wv] = br; }
            __syncthreads();
            if (tid == 0) {
                float fv = sWBestV[0]; int fr = sWBestR[0];
                for (int w2 = 1; w2 < NWAVES; ++w2) {
                    float ov = sWBestV[w2]; int orr = sWBestR[w2];
                    if (ov > fv || (ov == fv && orr < fr)) { fv = ov; fr = orr; }
                }
                sColMax[g2] = fv; sColArg[g2] = fr;
            }
            __syncthreads();
        }
    }
    __syncthreads();

    // ---- Phase 3: CIoU box loss + matched objectness logits ----
    const float* objRow = pred_obj + (size_t)b * KK;
    for (int i = tid; i < GG; i += NTHREADS) {
        if (sVs[i]) {
            float4 p = pb[sPs[i]];
            float4 g = sGt[sGs[i]];
            float tlx = fmaxf(p.x, g.x), tly = fmaxf(p.y, g.y);
            float brx = fminf(p.z, g.z), bry = fminf(p.w, g.w);
            float iw = fmaxf(brx - tlx, 0.f), ih = fmaxf(bry - tly, 0.f);
            float inter = iw * ih;
            float ap = areaOf(p.x, p.y, p.z, p.w);
            float ag = areaOf(g.x, g.y, g.z, g.w);
            float iou = inter / fmaxf(ap + ag - inter, EPSF);
            float px = (p.x + p.z) * 0.5f, py = (p.y + p.w) * 0.5f;
            float tx = (g.x + g.z) * 0.5f, ty = (g.y + g.w) * 0.5f;
            float rho2 = (px - tx) * (px - tx) + (py - ty) * (py - ty);
            float ex0 = fminf(p.x, g.x), ey0 = fminf(p.y, g.y);
            float ex1 = fmaxf(p.z, g.z), ey1 = fmaxf(p.w, g.w);
            float c2 = fmaxf((ex1 - ex0) * (ex1 - ex0) + (ey1 - ey0) * (ey1 - ey0), EPSF);
            float pw = fmaxf(p.z - p.x, EPSF), ph = fmaxf(p.w - p.y, EPSF);
            float tw = fmaxf(g.z - g.x, EPSF), th = fmaxf(g.w - g.y, EPSF);
            float dat = atanf(tw / th) - atanf(pw / ph);
            float v = (4.f / (3.14159265358979323846f * 3.14159265358979323846f)) * dat * dat;
            float alpha = v / (1.f - iou + v + EPSF);
            float loss = 1.f - (iou - rho2 / c2 - alpha * v);
            atomicAdd(&sLbox, loss);
            atomicAdd(&sObjNeg, objRow[sPs[i]]);
        }
    }

    // ---- Phase 4a: per-pair class-logit max + label logit ----
    for (int i = tid; i < GPAD; i += NTHREADS) {
        if (i < GG && sVs[i]) {
            const float* row = pred_cls + ((size_t)b * KK + sPs[i]) * CC;
            float m = -1e30f;
            for (int c = 0; c < CC; ++c) m = fmaxf(m, row[c]);
            sMaxl[i] = m;
            int lab = gt_labels[(size_t)b * GG + sGs[i]];
            sLabLogit[i] = row[lab];
        }
    }
    __syncthreads();

    // WMMA all-ones B matrix for row-sum reductions
    v16h ones;
    for (int h = 0; h < 16; ++h) ones[h] = (_Float16)1.0f;

    // ---- Phase 4b: sum-exp over classes via WMMA (16 pairs per wave-group) ----
    // Branchless operand build: unconditional clamped loads + multiplicative gate,
    // so loads batch behind a single s_wait and no EXEC-masked regions form.
    for (int gsel = wv; gsel < 13; gsel += NWAVES) {
        int base = gsel * 16;
        int pidx = base + (lane & 15);
        int prow = sPs[pidx];                       // 0 for invalid pairs (safe row)
        float gate0 = sVs[pidx] ? 1.f : 0.f;
        float ml = sMaxl[pidx];
        const float* row = pred_cls + ((size_t)b * KK + prow) * CC;
        v8f acc = {};
        for (int j = 0; j < 3; ++j) {
            int cbase = 32 * j + ((lane < 16) ? 0 : 16);
            float tmp[16];
#pragma unroll
            for (int h = 0; h < 16; ++h) {
                int c = cbase + h;
                tmp[h] = row[(c < CC) ? c : 0];
            }
            v16h a;
#pragma unroll
            for (int h = 0; h < 16; ++h) {
                int c = cbase + h;
                float gate = (c < CC) ? gate0 : 0.f;
                a[h] = (_Float16)(expf(tmp[h] - ml) * gate);
            }
            acc = __builtin_amdgcn_wmma_f32_16x16x32_f16(false, a, false, ones,
                                                         (short)0, acc, false, false);
        }
        // D layout: VGPR r -> M=r (lanes 0-15) / M=8+r (lanes 16-31)
        if (lane == 0)  { for (int r2 = 0; r2 < 8; ++r2) sSumexp[base + r2]     = acc[r2]; }
        if (lane == 16) { for (int r2 = 0; r2 < 8; ++r2) sSumexp[base + 8 + r2] = acc[r2]; }
    }
    __syncthreads();

    // ---- Phase 4c: cross-entropy sum ----
    for (int i = tid; i < GG; i += NTHREADS) {
        if (sVs[i]) {
            float lce = logf(sSumexp[i]) + sMaxl[i] - sLabLogit[i];
            atomicAdd(&sLcls, lce);
        }
    }

    // ---- Phase 5: softplus sum over all K objectness logits via WMMA ----
    {
        v8f acc = {};
        int base = wv * 512 + (lane & 15) * 32 + ((lane < 16) ? 0 : 16);
        v16h a;
#pragma unroll
        for (int h = 0; h < 16; ++h) {
            int idx = base + h;
            float x = objRow[(idx < KK) ? idx : 0];
            float s = logf(1.f + expf(-fabsf(x))) + fmaxf(x, 0.f);  // stable softplus
            a[h] = (_Float16)((idx < KK) ? s : 0.f);
        }
        acc = __builtin_amdgcn_wmma_f32_16x16x32_f16(false, a, false, ones,
                                                     (short)0, acc, false, false);
        float s = 0.f;
        for (int r2 = 0; r2 < 8; ++r2) s += acc[r2];   // half the row-sums per lane half
        s += __shfl_xor(s, 16);                        // combine both halves
        if (lane == 0) atomicAdd(&sObjPos, s);
    }
    __syncthreads();

    // ---- Final per-batch accumulation into global workspace ----
    if (tid == 0) {
        float n = (float)sN;
        float denom = fmaxf(n, 1.f);
        float hf = (sN > 0) ? 1.f : 0.f;
        atomicAdd(&ws[0], hf * sLbox / denom);
        atomicAdd(&ws[1], hf * sLcls / denom);
        atomicAdd(&ws[2], hf);
        atomicAdd(&ws[3], sObjPos - sObjNeg);
    }
}

__global__ void loss_final_kernel(const float* __restrict__ ws, float* __restrict__ out) {
    float nb = fmaxf(ws[2], 1.f);
    float lb = ws[0] / nb;
    float lc = ws[1] / nb;
    float lo = ws[3] / (float)(BB * KK);
    out[0] = 5.f * lb + 1.f * lc + 1.f * lo;
    out[1] = lb;
    out[2] = lc;
    out[3] = lo;
}

extern "C" void kernel_launch(void* const* d_in, const int* in_sizes, int n_in,
                              void* d_out, int out_size, void* d_ws, size_t ws_size,
                              hipStream_t stream) {
    (void)in_sizes; (void)n_in; (void)out_size; (void)ws_size;
    const float* pred_boxes = (const float*)d_in[0];
    const float* pred_obj   = (const float*)d_in[1];
    const float* pred_cls   = (const float*)d_in[2];
    const float* gt_boxes   = (const float*)d_in[3];
    const int*   gt_labels  = (const int*)d_in[4];
    const unsigned char* gt_mask = (const unsigned char*)d_in[5];  // jnp bool_ = 1 byte
    float* ws  = (float*)d_ws;
    float* out = (float*)d_out;

    loss_init_kernel<<<1, NTHREADS, 0, stream>>>(ws);
    loss_batch_kernel<<<BB, NTHREADS, 0, stream>>>(pred_boxes, pred_obj, pred_cls,
                                                   gt_boxes, gt_labels, gt_mask, ws);
    loss_final_kernel<<<1, 1, 0, stream>>>(ws, out);
}
